// Attention_15032385536099
// MI455X (gfx1250) — matrix operand order repaired and checked by reading, not compiled, not run on hardware
//
#include <hip/hip_runtime.h>

// ---------------------------------------------------------------------------
// PVT-style spatial-reduction attention for MI455X (gfx1250, wave32, WMMA).
// B=4, N=4096 (64x64), C=384, heads=6, qk_head_dim=32, head_dim=64, SR=2,
// Nk=1024, scale=1/8. All GEMMs + attention core use v_wmma_f32_16x16x32_bf16.
// The softmax scale (0.125*log2e) is folded into the Wq bf16 conversion, so
// scores come out of the QK^T WMMA already in the exp2 domain. Row-maxes use
// immediate-pattern ds_swizzle butterflies with raw v_max_num_f32; row-sums
// come from an extra WMMA against an all-ones column block.
// ---------------------------------------------------------------------------

typedef __attribute__((ext_vector_type(8)))  __bf16 v8bf;
typedef __attribute__((ext_vector_type(16))) __bf16 v16bf;
typedef __attribute__((ext_vector_type(8)))  float  v8f;

#define B_    4
#define NTOK  4096
#define C_    384
#define HEADS 6
#define QKD   32
#define HD    64
#define NK    1024
// 0.125 * log2(e): folded into Wq so S = (q*Wq_scaled)·k is log2-domain.
#define SC_LOG2E 0.1803368801111244f

__device__ __forceinline__ v8f wmma_bf16(v16bf a, v16bf b, v8f c) {
  return __builtin_amdgcn_wmma_f32_16x16x32_bf16(
      /*neg_a=*/false, a, /*neg_b=*/false, b,
      /*c_mod=*/(short)0, c, /*reuse_a=*/false, /*reuse_b=*/false);
}

// Raw max (skips the canonicalize the compiler adds around fmaxf of swizzled
// bits). Pure VALU asm: register constraints carry all dependencies.
__device__ __forceinline__ float maxraw(float a, float b) {
  float d;
  asm("v_max_num_f32 %0, %1, %2" : "=v"(d) : "v"(a), "v"(b));
  return d;
}

// Immediate-pattern XOR swizzle across 32 lanes (and_mask=0x1F, or=0).
#define SWZ_MAXF(t, pat)                                                      \
  (t) = maxraw((t), __int_as_float(                                           \
            __builtin_amdgcn_ds_swizzle(__float_as_int(t), (pat))))
#define SWZ_X1 0x041F
#define SWZ_X2 0x081F
#define SWZ_X4 0x101F
#define SWZ_X8 0x201F

// B-fragment (32x16, K x N): lane = col (lane&15), 16 contiguous K values at
// +16*(lane>>4). Caller passes pointer to the 16 contiguous bf16 elements.
__device__ __forceinline__ v16bf ld16(const __bf16* p) {
  v8bf lo = *(const v8bf*)p;
  v8bf hi = *(const v8bf*)(p + 8);
  return __builtin_shufflevector(lo, hi, 0,1,2,3,4,5,6,7,8,9,10,11,12,13,14,15);
}

// A-fragment (16x32, M x K): lane row = lane&15, kh = lane>>4.
// elements 0..15 = K[16*kh .. 16*kh+15] (contiguous).
__device__ __forceinline__ v16bf ld_afrag(const __bf16* row, int kh) {
  v8bf lo = *(const v8bf*)(row + 16 * kh);
  v8bf hi = *(const v8bf*)(row + 16 * kh + 8);
  return __builtin_shufflevector(lo, hi, 0,1,2,3,4,5,6,7,8,9,10,11,12,13,14,15);
}

// ---------------------------------------------------------------------------
// f32 -> bf16 conversion with optional constant scale
// ---------------------------------------------------------------------------
__global__ void cvt_f32_bf16(const float* __restrict__ in,
                             __bf16* __restrict__ out, int n, float scale) {
  int i = blockIdx.x * blockDim.x + threadIdx.x;
  if (i < n) out[i] = (__bf16)(in[i] * scale);
}

// ---------------------------------------------------------------------------
// Depthwise 2x2/s2 conv + bias + BatchNorm(eval) -> bf16 x_sr (B*NK, C)
// ---------------------------------------------------------------------------
__global__ void sr_conv_bn(const float* __restrict__ x,
                           const float* __restrict__ srw,
                           const float* __restrict__ srb,
                           const float* __restrict__ gamma,
                           const float* __restrict__ beta,
                           const float* __restrict__ mean,
                           const float* __restrict__ var,
                           __bf16* __restrict__ out) {
  int idx = blockIdx.x * blockDim.x + threadIdx.x;
  if (idx >= B_ * NK * C_) return;
  int c  = idx % C_;
  int t  = idx / C_;
  int nk = t % NK;
  int b  = t / NK;
  int oh = nk >> 5, ow = nk & 31;
  int r0 = (2 * oh) * 64 + 2 * ow;
  const float* xb = x + (size_t)b * NTOK * C_ + c;
  float a = xb[(size_t)r0 * C_]        * srw[c * 4 + 0]
          + xb[(size_t)(r0 + 1) * C_]  * srw[c * 4 + 1]
          + xb[(size_t)(r0 + 64) * C_] * srw[c * 4 + 2]
          + xb[(size_t)(r0 + 65) * C_] * srw[c * 4 + 3];
  a += srb[c];
  float inv = gamma[c] * rsqrtf(var[c] + 1e-5f);
  a = (a - mean[c]) * inv + beta[c];
  out[idx] = (__bf16)a;
}

// ---------------------------------------------------------------------------
// Generic WMMA GEMM: out(M,N) = A(M,K) @ W(N,K)^T [+ bias]
// A row-major bf16, W torch-layout (N,K) bf16 (== natural B-fragment source).
// Each wave computes a 16x64 tile: 1 A-frag reused over 4 B-frags per K-step.
// OMODE: 0 = f32 row-major, 1 = bf16 row-major, 2 = bf16 col-major (out[n*M+m])
// ---------------------------------------------------------------------------
template <int OMODE, bool BIAS>
__global__ void __launch_bounds__(256)
gemm_bf16(const __bf16* __restrict__ A, const __bf16* __restrict__ W,
          const float* __restrict__ bias, void* __restrict__ outv,
          int M, int N, int K, int Mtiles) {
  int lane = threadIdx.x & 31;
  int wid  = blockIdx.x * (blockDim.x >> 5) + (threadIdx.x >> 5);
  int mt = wid % Mtiles, nt = wid / Mtiles;
  int m0 = mt * 16, n0 = nt * 64;
  int c = lane & 15, hh = lane >> 4;

  v8f acc[4] = {v8f{}, v8f{}, v8f{}, v8f{}};
  const __bf16* arow = A + (size_t)(m0 + c) * K;
  const __bf16* w0   = W + (size_t)(n0 + c) * K + 16 * hh;

  for (int k0 = 0; k0 < K; k0 += 32) {
    v16bf a = ld_afrag(arow + k0, hh);
    v16bf b0 = ld16(w0 + k0);
    v16bf b1 = ld16(w0 + (size_t)16 * K + k0);
    v16bf b2 = ld16(w0 + (size_t)32 * K + k0);
    v16bf b3 = ld16(w0 + (size_t)48 * K + k0);
    acc[0] = wmma_bf16(a, b0, acc[0]);
    acc[1] = wmma_bf16(a, b1, acc[1]);
    acc[2] = wmma_bf16(a, b2, acc[2]);
    acc[3] = wmma_bf16(a, b3, acc[3]);
  }

#pragma unroll
  for (int r = 0; r < 8; ++r) {
    int row = m0 + r + 8 * hh;
#pragma unroll
    for (int j = 0; j < 4; ++j) {
      int col = n0 + 16 * j + c;
      float v = acc[j][r];
      if (BIAS) v += bias[col];
      if (OMODE == 0)
        ((float*)outv)[(size_t)row * N + col] = v;
      else if (OMODE == 1)
        ((__bf16*)outv)[(size_t)row * N + col] = (__bf16)v;
      else
        ((__bf16*)outv)[(size_t)col * M + row] = (__bf16)v;
    }
  }
}

// ---------------------------------------------------------------------------
// Fused flash attention. One wave per (b, head, 16-query tile).
//   q  : (B*NTOK, 192) bf16 row-major, PRE-SCALED by 0.125*log2e
//   k  : (B*NK,   192) bf16 row-major
//   vt : (384, B*NK)   bf16  == v^T, row = h*64+d, col = b*1024+nk
//   ao : (B*NTOK, 384) bf16 attention output (pre out-proj)
// Per 32-key chunk: 2 WMMAs (S, log2 domain), online max via ds_swizzle
// butterflies, P restaged through per-wave LDS into an A-fragment, 4 WMMAs
// (P@V) plus 1 WMMA against an all-ones block for the softmax denominators.
// ---------------------------------------------------------------------------
__global__ void __launch_bounds__(256)
attn_flash(const __bf16* __restrict__ q, const __bf16* __restrict__ k,
           const __bf16* __restrict__ vt, __bf16* __restrict__ ao) {
  __shared__ __bf16 plds[8][16][32];   // per-wave 16x32 P staging (8 KB total)

  int lane = threadIdx.x & 31;
  int w    = threadIdx.x >> 5;
  int wid  = blockIdx.x * 8 + w;
  int mt = wid & 255;                  // 256 query tiles of 16
  int h  = (wid >> 8) % HEADS;
  int b  = wid / (256 * HEADS);
  int c = lane & 15, hh = lane >> 4;
  int m0 = mt * 16;

  // Q fragment: row = query token, K = 32 head dims (one fragment, reused).
  const __bf16* qrow = q + (size_t)(b * NTOK + m0 + c) * 192 + h * QKD;
  v16bf qa = ld_afrag(qrow, hh);

  // Constant all-ones B-fragment: row-sum side-car for the denominators.
  v16bf vones;
#pragma unroll
  for (int i = 0; i < 16; ++i) vones[i] = (__bf16)1.0f;

  v8f acc[4] = {v8f{}, v8f{}, v8f{}, v8f{}};
  v8f accl{};                          // softmax denominators (P @ ones)
  float mrow[8];
#pragma unroll
  for (int r = 0; r < 8; ++r) mrow[r] = -1e30f;

  const __bf16* kbase = k  + (size_t)(b * NK + c) * 192 + h * QKD + 16 * hh;
  const __bf16* vbase = vt + (size_t)(h * HD + c) * (B_ * NK) + b * NK + 16 * hh;

  for (int n0 = 0; n0 < NK; n0 += 32) {
    // S = q @ k^T for keys [n0, n0+32); q pre-scaled -> S is log2-domain.
    v16bf kb0 = ld16(kbase + (size_t)n0 * 192);
    v16bf kb1 = ld16(kbase + (size_t)(n0 + 16) * 192);
    v8f z{};
    v8f s0 = wmma_bf16(qa, kb0, z);
    v8f s1 = wmma_bf16(qa, kb1, z);

    // Online softmax. C-fragment: lane holds 8 rows (r + 8*hh), col = c.
    // Row max = XOR butterflies across the 16-lane half (immediate swizzle).
#pragma unroll
    for (int r = 0; r < 8; ++r) {
      float t0 = s0[r], t1 = s1[r];
      float t = maxraw(t0, t1);
      SWZ_MAXF(t, SWZ_X1);
      SWZ_MAXF(t, SWZ_X2);
      SWZ_MAXF(t, SWZ_X4);
      SWZ_MAXF(t, SWZ_X8);
      float mn   = maxraw(mrow[r], t);
      float corr = exp2f(mrow[r] - mn);
      mrow[r] = mn;
      float p0 = exp2f(t0 - mn);
      float p1 = exp2f(t1 - mn);
      acc[0][r] *= corr; acc[1][r] *= corr;
      acc[2][r] *= corr; acc[3][r] *= corr;
      accl[r]   *= corr;
      plds[w][r + 8 * hh][c]      = (__bf16)p0;
      plds[w][r + 8 * hh][16 + c] = (__bf16)p1;
    }

    // Intra-wave LDS is in-order; make the reordering constraint explicit.
    asm volatile("s_wait_dscnt 0x0" ::: "memory");

    // Re-load P as a 16x32 A-fragment (row = query = c, kh = hh).
    v8bf plo = *(const v8bf*)&plds[w][c][16 * hh];
    v8bf phi = *(const v8bf*)&plds[w][c][16 * hh + 8];
    v16bf pa = __builtin_shufflevector(plo, phi,
                 0,1,2,3,4,5,6,7,8,9,10,11,12,13,14,15);

    // V fragments from v^T: contiguous 16 keys per lane, col = head dim.
    v16bf vb0 = ld16(vbase + n0);
    v16bf vb1 = ld16(vbase + (size_t)16 * (B_ * NK) + n0);
    v16bf vb2 = ld16(vbase + (size_t)32 * (B_ * NK) + n0);
    v16bf vb3 = ld16(vbase + (size_t)48 * (B_ * NK) + n0);
    acc[0] = wmma_bf16(pa, vb0, acc[0]);
    acc[1] = wmma_bf16(pa, vb1, acc[1]);
    acc[2] = wmma_bf16(pa, vb2, acc[2]);
    acc[3] = wmma_bf16(pa, vb3, acc[3]);
    accl   = wmma_bf16(pa, vones, accl);   // denominator side-car
  }

  // Normalize and store (B*NTOK, 384) bf16, head h at cols h*64..h*64+63.
#pragma unroll
  for (int r = 0; r < 8; ++r) {
    float inv = 1.0f / accl[r];
    __bf16* o = ao + (size_t)(b * NTOK + m0 + r + 8 * hh) * C_ + h * HD + c;
    o[0]  = (__bf16)(acc[0][r] * inv);
    o[16] = (__bf16)(acc[1][r] * inv);
    o[32] = (__bf16)(acc[2][r] * inv);
    o[48] = (__bf16)(acc[3][r] * inv);
  }
}

// ---------------------------------------------------------------------------
// Host-side orchestration (graph-capture safe: only kernel launches).
// ---------------------------------------------------------------------------
extern "C" void kernel_launch(void* const* d_in, const int* in_sizes, int n_in,
                              void* d_out, int out_size, void* d_ws,
                              size_t ws_size, hipStream_t stream) {
  (void)in_sizes; (void)n_in; (void)out_size; (void)ws_size;
  const float* x    = (const float*)d_in[0];
  // d_in[1]=H, d_in[2]=W (64/64, baked into the kernels)
  const float* Wq   = (const float*)d_in[3];
  const float* Wk   = (const float*)d_in[4];
  const float* Wv   = (const float*)d_in[5];
  const float* srw  = (const float*)d_in[6];
  const float* srb  = (const float*)d_in[7];
  const float* g    = (const float*)d_in[8];
  const float* bt   = (const float*)d_in[9];
  const float* mean = (const float*)d_in[10];
  const float* var  = (const float*)d_in[11];
  const float* Wp   = (const float*)d_in[12];
  const float* bp   = (const float*)d_in[13];
  float* out = (float*)d_out;

  char* ws = (char*)d_ws;
  auto carve = [&](size_t bytes) {
    char* p = ws;
    ws += (bytes + 255) & ~(size_t)255;
    return p;
  };
  __bf16* x_bf   = (__bf16*)carve((size_t)B_ * NTOK * C_ * 2);   // 12.6 MB
  __bf16* xsr_bf = (__bf16*)carve((size_t)B_ * NK * C_ * 2);     //  3.1 MB
  __bf16* q_bf   = (__bf16*)carve((size_t)B_ * NTOK * 192 * 2);  //  6.3 MB
  __bf16* k_bf   = (__bf16*)carve((size_t)B_ * NK * 192 * 2);    //  1.6 MB
  __bf16* vt_bf  = (__bf16*)carve((size_t)C_ * B_ * NK * 2);     //  3.1 MB
  __bf16* ao_bf  = (__bf16*)carve((size_t)B_ * NTOK * C_ * 2);   // 12.6 MB
  __bf16* wq_bf  = (__bf16*)carve((size_t)192 * C_ * 2);
  __bf16* wk_bf  = (__bf16*)carve((size_t)192 * C_ * 2);
  __bf16* wv_bf  = (__bf16*)carve((size_t)C_ * C_ * 2);
  __bf16* wp_bf  = (__bf16*)carve((size_t)C_ * C_ * 2);

  // 1) precision conversion; Wq carries the softmax scale in log2 domain.
  cvt_f32_bf16<<<24576, 256, 0, stream>>>(x, x_bf, B_ * NTOK * C_, 1.0f);
  cvt_f32_bf16<<<288,   256, 0, stream>>>(Wq, wq_bf, 192 * C_, SC_LOG2E);
  cvt_f32_bf16<<<288,   256, 0, stream>>>(Wk, wk_bf, 192 * C_, 1.0f);
  cvt_f32_bf16<<<576,   256, 0, stream>>>(Wv, wv_bf, C_ * C_, 1.0f);
  cvt_f32_bf16<<<576,   256, 0, stream>>>(Wp, wp_bf, C_ * C_, 1.0f);

  // 2) spatial reduction (conv 2x2/s2 + bias + BN) -> bf16
  sr_conv_bn<<<6144, 256, 0, stream>>>(x, srw, srb, g, bt, mean, var, xsr_bf);

  // 3) projections (WMMA GEMMs). waves = Mtiles * (N/64); blocks = waves/8.
  gemm_bf16<1, false><<<384, 256, 0, stream>>>(x_bf, wq_bf, nullptr, q_bf,
                                               B_ * NTOK, 192, C_, 1024);
  gemm_bf16<1, false><<<96, 256, 0, stream>>>(xsr_bf, wk_bf, nullptr, k_bf,
                                              B_ * NK, 192, C_, 256);
  gemm_bf16<2, false><<<192, 256, 0, stream>>>(xsr_bf, wv_bf, nullptr, vt_bf,
                                               B_ * NK, C_, C_, 256);

  // 4) fused flash attention: 4*6*256 = 6144 waves
  attn_flash<<<768, 256, 0, stream>>>(q_bf, k_bf, vt_bf, ao_bf);

  // 5) output projection + bias -> f32 d_out
  gemm_bf16<0, true><<<768, 256, 0, stream>>>(ao_bf, wp_bf, bp, out,
                                              B_ * NTOK, C_, C_, 1024);
}
